// MoEGate_86655260164506
// MI455X (gfx1250) — matrix operand order repaired and checked by reading, not compiled, not run on hardware
//
#include <hip/hip_runtime.h>
#include <hip/hip_bf16.h>

typedef __bf16 bf16_t;
typedef __attribute__((ext_vector_type(16))) __bf16 v16bf;
typedef __attribute__((ext_vector_type(4)))  __bf16 v4bf;
typedef __attribute__((ext_vector_type(8)))  float  v8f;

#define H_DIM 4096
#define N_EXPERTS 64
#define TOP_K 2
#define TOKENS_PER_WAVE 16
#define WAVES_PER_BLOCK 8
#define TOKENS_PER_BLOCK (TOKENS_PER_WAVE * WAVES_PER_BLOCK)

// ---------------------------------------------------------------------------
// Phase 0: convert router weight fp32 -> bf16 (0.5 MB, stays L2-resident).
// ---------------------------------------------------------------------------
__global__ void convert_weight_bf16(const float* __restrict__ w,
                                    bf16_t* __restrict__ wb, int n4) {
    int i = blockIdx.x * blockDim.x + threadIdx.x;
    if (i < n4) {
        float4 v = ((const float4*)w)[i];
        v4bf o;
        o[0] = (bf16_t)v.x; o[1] = (bf16_t)v.y;
        o[2] = (bf16_t)v.z; o[3] = (bf16_t)v.w;
        ((v4bf*)wb)[i] = o;
    }
}

// ---------------------------------------------------------------------------
// Phase 1+2: per wave, 16 tokens x 64 experts logits via bf16 WMMA, then
// noise + softmax + top-2 through LDS.
// ---------------------------------------------------------------------------
__global__ __launch_bounds__(256) void moe_gate_kernel(
    const float*  __restrict__ x,       // [N, H] fp32
    const bf16_t* __restrict__ wb,      // [E, H] bf16
    const float*  __restrict__ noise,   // [N, E] fp32
    int*          __restrict__ out_idx, // [N, 2] int32
    float*        __restrict__ out_w,   // [N, 2] fp32
    int n_tokens)
{
    __shared__ float lds[WAVES_PER_BLOCK][TOKENS_PER_WAVE][N_EXPERTS + 1];

    const int lane = threadIdx.x & 31;
    const int wave = threadIdx.x >> 5;
    const int tokBase = blockIdx.x * TOKENS_PER_BLOCK + wave * TOKENS_PER_WAVE;

    const int m     = lane & 15;   // token row (A) / expert col (B,C)
    const int hi    = lane >> 4;   // lane half selector
    const int koff  = hi * 8;      // unused; A fragment uses khalf instead
    const int khalf = hi * 16;     // A/B K half: this code's contiguous k = 16*hi + s

    const float* xrow = x + (size_t)(tokBase + m) * H_DIM;
    const bf16_t* w0 = wb + (size_t)( 0 + m) * H_DIM + khalf;
    const bf16_t* w1 = wb + (size_t)(16 + m) * H_DIM + khalf;
    const bf16_t* w2 = wb + (size_t)(32 + m) * H_DIM + khalf;
    const bf16_t* w3 = wb + (size_t)(48 + m) * H_DIM + khalf;

    v8f c0 = {}, c1 = {}, c2 = {}, c3 = {};

    for (int k0 = 0; k0 < H_DIM; k0 += 32) {
        // ---- A fragment: 16 tokens x 32 K, fp32 load -> bf16 regs ----
        // lanes 0-15 hold K = k0+0..15 for token M = lane
        // lanes 16-31 hold K = k0+16..31
        float4 a0 = *(const float4*)(xrow + k0 + khalf);
        float4 a1 = *(const float4*)(xrow + k0 + khalf + 4);
        float4 a2 = *(const float4*)(xrow + k0 + khalf + 8);
        float4 a3 = *(const float4*)(xrow + k0 + khalf + 12);
        v16bf A;
        A[0]  = (bf16_t)a0.x; A[1]  = (bf16_t)a0.y;
        A[2]  = (bf16_t)a0.z; A[3]  = (bf16_t)a0.w;
        A[4]  = (bf16_t)a1.x; A[5]  = (bf16_t)a1.y;
        A[6]  = (bf16_t)a1.z; A[7]  = (bf16_t)a1.w;
        A[8]  = (bf16_t)a2.x; A[9]  = (bf16_t)a2.y;
        A[10] = (bf16_t)a2.z; A[11] = (bf16_t)a2.w;
        A[12] = (bf16_t)a3.x; A[13] = (bf16_t)a3.y;
        A[14] = (bf16_t)a3.z; A[15] = (bf16_t)a3.w;

        // ---- B fragments: 32 K x 16 experts, contiguous 32B per lane ----
        v16bf B0 = *(const v16bf*)(w0 + k0);
        v16bf B1 = *(const v16bf*)(w1 + k0);
        v16bf B2 = *(const v16bf*)(w2 + k0);
        v16bf B3 = *(const v16bf*)(w3 + k0);

        c0 = __builtin_amdgcn_wmma_f32_16x16x32_bf16(false, A, false, B0,
                                                     (short)0, c0, false, false);
        c1 = __builtin_amdgcn_wmma_f32_16x16x32_bf16(false, A, false, B1,
                                                     (short)0, c1, false, false);
        c2 = __builtin_amdgcn_wmma_f32_16x16x32_bf16(false, A, false, B2,
                                                     (short)0, c2, false, false);
        c3 = __builtin_amdgcn_wmma_f32_16x16x32_bf16(false, A, false, B3,
                                                     (short)0, c3, false, false);
    }

    // ---- scatter C + noise into LDS (C layout: M = r + 8*hi, N = lane&15) ----
    v8f cc[4] = {c0, c1, c2, c3};
    const int eCol = lane & 15;
#pragma unroll
    for (int t = 0; t < 4; ++t) {
#pragma unroll
        for (int r = 0; r < 8; ++r) {
            int M = r + 8 * hi;
            int e = t * 16 + eCol;
            float v = cc[t][r] + noise[(size_t)(tokBase + M) * N_EXPERTS + e];
            lds[wave][M][e] = v;
        }
    }

    __syncthreads();

    // ---- per-token softmax + top-2 (lanes 0-15 each own one token) ----
    if ((lane < TOKENS_PER_WAVE) && (tokBase + lane < n_tokens)) {
        const int tok = tokBase + lane;
        const float* row = &lds[wave][lane][0];

        float m1 = -3.402823466e38f, m2 = -3.402823466e38f;
        int i1 = 0, i2 = 0;
#pragma unroll 4
        for (int e = 0; e < N_EXPERTS; ++e) {
            float v = row[e];
            if (v > m1) { m2 = m1; i2 = i1; m1 = v; i1 = e; }
            else if (v > m2) { m2 = v; i2 = e; }
        }
        float s = 0.0f;
#pragma unroll 4
        for (int e = 0; e < N_EXPERTS; ++e)
            s += __expf(row[e] - m1);
        float inv = 1.0f / s;

        out_idx[tok * TOP_K + 0] = i1;
        out_idx[tok * TOP_K + 1] = i2;
        out_w[tok * TOP_K + 0] = inv;                    // exp(m1-m1)/s
        out_w[tok * TOP_K + 1] = __expf(m2 - m1) * inv;  // exp(m2-m1)/s
    }
}

extern "C" void kernel_launch(void* const* d_in, const int* in_sizes, int n_in,
                              void* d_out, int out_size, void* d_ws, size_t ws_size,
                              hipStream_t stream) {
    const float* x     = (const float*)d_in[0];   // [B,S,H] fp32
    const float* w     = (const float*)d_in[1];   // [E,H]   fp32
    const float* noise = (const float*)d_in[2];   // [N,E]   fp32

    const int n_tokens = in_sizes[2] / N_EXPERTS;     // 16384
    const int wn       = in_sizes[1];                 // E*H

    bf16_t* wb = (bf16_t*)d_ws;                       // 512 KB bf16 weight

    // outputs: [N,2] int32 indices, then [N,2] fp32 weights
    int*   out_idx = (int*)d_out;
    float* out_w   = (float*)d_out + (size_t)n_tokens * TOP_K;

    int n4 = wn / 4;
    convert_weight_bf16<<<(n4 + 255) / 256, 256, 0, stream>>>(w, wb, n4);

    dim3 grid(n_tokens / TOKENS_PER_BLOCK);
    moe_gate_kernel<<<grid, 256, 0, stream>>>(x, wb, noise, out_idx, out_w,
                                              n_tokens);
}
